// CodebookModule_75342316306508
// MI455X (gfx1250) — matrix operand order repaired and checked by reading, not compiled, not run on hardware
//
#include <hip/hip_runtime.h>
#include <stdint.h>

typedef __attribute__((ext_vector_type(16))) _Float16 v16h;
typedef __attribute__((ext_vector_type(8)))  _Float16 v8h;
typedef __attribute__((ext_vector_type(8)))  float    v8f;
typedef __attribute__((__vector_size__(16))) int      gv4i;   // matches builtin param type

#define N_ROWS 32768   // B*L = 64*512
#define DIM    512
#define KCB    1024

#define FM     32      // rows per block in fused kernel
#define ASTRF  520     // A LDS stride (f16): 512 data + 8 pad (1040B, 16B aligned)
#define BSTR   40      // B LDS stride (f16): 32 data + 8 pad (80B, 16B aligned)

#if defined(__gfx1250__) && __has_builtin(__builtin_amdgcn_global_load_async_to_lds_b128)
#define HAVE_ASYNC_LDS 1
#endif

// async global->LDS 16B copy (CDNA5 path, ASYNCcnt-tracked); sync fallback
__device__ __forceinline__ void cp_b128_to_lds(const _Float16* g, _Float16* l) {
#ifdef HAVE_ASYNC_LDS
  __builtin_amdgcn_global_load_async_to_lds_b128(
      (__attribute__((address_space(1))) gv4i*)(void*)g,
      (__attribute__((address_space(3))) gv4i*)(void*)l, 0, 0);
#else
  *(v8h*)l = *(const v8h*)g;
#endif
}
__device__ __forceinline__ void async_wait_all() {
#ifdef HAVE_ASYNC_LDS
#if __has_builtin(__builtin_amdgcn_s_wait_asynccnt)
  __builtin_amdgcn_s_wait_asynccnt(0);
#else
  asm volatile("s_wait_asynccnt 0" ::: "memory");
#endif
#endif
}

// ---------------- wave reductions (wave32) ----------------
__device__ __forceinline__ float wave_sum(float v) {
#pragma unroll
  for (int off = 16; off; off >>= 1) v += __shfl_xor(v, off, 32);
  return v;
}

// counter-based gumbel noise (stateless hash RNG)
__device__ __forceinline__ float gumbel_noise(uint32_t row, uint32_t k, uint32_t seed) {
  uint32_t x = row * 1024u + k;
  x ^= seed;
  x ^= x >> 16; x *= 0x7feb352dU;
  x ^= x >> 15; x *= 0x846ca68bU;
  x ^= x >> 16;
  float u = (float)(x >> 8) * (1.0f / 16777216.0f) + 5.9604645e-8f; // (0,1)
  return -__logf(-__logf(u));
}

// ---------------- kernel 1: mixed normalized embedding -> f16 ----------------
// m = 0.5 * s/max(||s||,eps) + 0.5 * a/max(||a||,eps)
__global__ __launch_bounds__(256) void mix_norm_kernel(const float* __restrict__ S,
                                                       const float* __restrict__ A,
                                                       _Float16* __restrict__ M) {
  const int lane = threadIdx.x & 31;
  const int row  = blockIdx.x * 8 + (threadIdx.x >> 5);
  const float* sr = S + (size_t)row * DIM;
  const float* ar = A + (size_t)row * DIM;
  float sv[16], av[16];
  float ss = 0.f, sa = 0.f;
#pragma unroll
  for (int j = 0; j < 16; j++) {
    sv[j] = sr[j * 32 + lane]; ss += sv[j] * sv[j];
    av[j] = ar[j * 32 + lane]; sa += av[j] * av[j];
  }
  ss = wave_sum(ss);
  sa = wave_sum(sa);
  const float ks = 0.5f / fmaxf(__fsqrt_rn(ss), 1e-8f);
  const float ka = 0.5f / fmaxf(__fsqrt_rn(sa), 1e-8f);
  _Float16* mr = M + (size_t)row * DIM;
#pragma unroll
  for (int j = 0; j < 16; j++)
    mr[j * 32 + lane] = (_Float16)(sv[j] * ks + av[j] * ka);
}

// ---------------- kernel 2: normalized codebook -> f16 ----------------
__global__ __launch_bounds__(256) void cb_norm_kernel(const float* __restrict__ CB,
                                                      _Float16* __restrict__ CN) {
  const int lane = threadIdx.x & 31;
  const int row  = blockIdx.x * 8 + (threadIdx.x >> 5);
  const float* cr = CB + (size_t)row * DIM;
  float cv[16];
  float ss = 0.f;
#pragma unroll
  for (int j = 0; j < 16; j++) { cv[j] = cr[j * 32 + lane]; ss += cv[j] * cv[j]; }
  ss = wave_sum(ss);
  const float kinv = 1.0f / fmaxf(__fsqrt_rn(ss), 1e-8f);
  _Float16* cn = CN + (size_t)row * DIM;
#pragma unroll
  for (int j = 0; j < 16; j++)
    cn[j * 32 + lane] = (_Float16)(cv[j] * kinv);
}

// ---------------- fused kernel: WMMA GEMM + dual gumbel softmax + gather ----
// Block: 32 rows x full K=1024 cols. 8 waves: wm in {0,1} (16-row half),
// wn in {0..3} (256-col slab, 16 WMMA tiles per wave). sim stays in VGPRs.
__global__ __launch_bounds__(256) void sim_gumbel_fused(
    const _Float16* __restrict__ A,      // [N_ROWS, DIM] f16 (mixed normalized)
    const _Float16* __restrict__ Bm,     // [KCB, DIM] f16 (normalized codebook)
    const float* __restrict__ codebook,  // [KCB, DIM] f32 (raw, for z_q gather)
    float* __restrict__ zq,              // [N_ROWS, DIM]
    float* __restrict__ wsoft,           // [N_ROWS, KCB]
    float* __restrict__ idxo) {          // [N_ROWS]
  __shared__ _Float16 Asf[FM * ASTRF];        // ~33 KB
  __shared__ _Float16 Bsf[KCB * BSTR];        // ~80 KB
  __shared__ float s_rv[FM][4];
  __shared__ int   s_ri[FM][4];
  __shared__ float s_stat[FM];
  __shared__ int   s_idx[FM];

  const int tid  = threadIdx.x;
  const int lane = tid & 31;
  const int w    = tid >> 5;
  const int wm   = w >> 2;        // 0..1
  const int wn   = w & 3;         // 0..3
  const int lm   = lane & 15;
  const int half = lane >> 4;
  const int m0   = blockIdx.x * FM;

  // ---- stage A slab (32 x 512 f16) into LDS once (async) ----
  {
    const int ar = tid >> 3;             // 0..31
    const int ac = (tid & 7) * 64;       // 0,64,...,448
    const _Float16* src = A + (size_t)(m0 + ar) * DIM + ac;
    _Float16* dst = Asf + ar * ASTRF + ac;
#pragma unroll
    for (int q = 0; q < 8; q++) cp_b128_to_lds(src + q * 8, dst + q * 8);
  }

  v8f acc[16];
#pragma unroll
  for (int j = 0; j < 16; j++) acc[j] = (v8f){};

  const int brow = tid >> 1;             // 0..127
  const int boff = (tid & 1) * 16;       // 0 or 16

  // ---- k loop: stage B tile (1024 x 32 f16) async, 16 WMMAs per wave ----
  // unroll 1: keep VGPRs < 256 (avoid s_set_vgpr_msb, preserve occupancy);
  // the 16-WMMA inner loop already gives the scheduler plenty of ILP.
#pragma unroll 1
  for (int kc = 0; kc < DIM; kc += 32) {
#pragma unroll
    for (int q = 0; q < 8; q++) {
      const int n = q * 128 + brow;      // covers all 1024 rows
      const _Float16* src = Bm + (size_t)n * DIM + kc + boff;
      _Float16* dst = Bsf + n * BSTR + boff;
      cp_b128_to_lds(src, dst);
      cp_b128_to_lds(src + 8, dst + 8);
      // pull next chunk's line toward the WGP while WMMAs run
      if (kc + 32 < DIM) __builtin_prefetch(src + 32, 0, 3);
    }
    async_wait_all();
    __syncthreads();

    // A fragment: lane m = lm, K pattern per ISA 16-bit A layout
    union { v16h v; v8h h[2]; } ua;
    const _Float16* pa = Asf + (wm * 16 + lm) * ASTRF + kc;
    ua.h[0] = *(const v8h*)(pa + half * 8);
    ua.h[1] = *(const v8h*)(pa + 16 + half * 8);
    const v16h afr = ua.v;

#pragma unroll
    for (int j = 0; j < 16; j++) {
      const _Float16* pb = Bsf + (wn * 256 + j * 16 + lm) * BSTR;
      union { v16h v; v8h h[2]; } ub;
      ub.h[0] = *(const v8h*)(pb + half * 8);
      ub.h[1] = *(const v8h*)(pb + 16 + half * 8);
      acc[j] = __builtin_amdgcn_wmma_f32_16x16x32_f16(
          false, afr, false, ub.v, (short)0, acc[j], false, false);
    }
    __syncthreads();
  }

  // C layout: VGPR r -> local row (wm*16 + half*8 + r), col = wn*256 + j*16 + lm
  const int lrow0 = wm * 16 + half * 8;

  // ---- pass 1: hard path, argmax(sim + g1) per row ----
  float bestv[8]; int besti[8];
#pragma unroll
  for (int r = 0; r < 8; r++) { bestv[r] = -3.4e38f; besti[r] = 0; }
#pragma unroll
  for (int j = 0; j < 16; j++) {
    const int col = wn * 256 + j * 16 + lm;
#pragma unroll
    for (int r = 0; r < 8; r++) {
      const float y = acc[j][r] + gumbel_noise(m0 + lrow0 + r, col, 0x2545F491u);
      if (y > bestv[r]) { bestv[r] = y; besti[r] = col; }
    }
  }
#pragma unroll
  for (int r = 0; r < 8; r++) {
#pragma unroll
    for (int off = 8; off; off >>= 1) {
      const float ov = __shfl_xor(bestv[r], off, 32);
      const int   oi = __shfl_xor(besti[r], off, 32);
      if (ov > bestv[r] || (ov == bestv[r] && oi < besti[r])) { bestv[r] = ov; besti[r] = oi; }
    }
    if (lm == 0) { s_rv[lrow0 + r][wn] = bestv[r]; s_ri[lrow0 + r][wn] = besti[r]; }
  }
  __syncthreads();
  if (tid < FM) {
    float bv = s_rv[tid][0]; int bix = s_ri[tid][0];
#pragma unroll
    for (int q = 1; q < 4; q++)
      if (s_rv[tid][q] > bv || (s_rv[tid][q] == bv && s_ri[tid][q] < bix)) {
        bv = s_rv[tid][q]; bix = s_ri[tid][q];
      }
    s_idx[tid] = bix;
    idxo[m0 + tid] = (float)bix;
  }
  __syncthreads();

  // ---- pass 2: soft path, y = sim + g2; row max ----
  float mx[8];
#pragma unroll
  for (int r = 0; r < 8; r++) mx[r] = -3.4e38f;
#pragma unroll
  for (int j = 0; j < 16; j++) {
    const int col = wn * 256 + j * 16 + lm;
#pragma unroll
    for (int r = 0; r < 8; r++) {
      const float y = acc[j][r] + gumbel_noise(m0 + lrow0 + r, col, 0x9E3779B9u);
      acc[j][r] = y;
      mx[r] = fmaxf(mx[r], y);
    }
  }
#pragma unroll
  for (int r = 0; r < 8; r++) {
#pragma unroll
    for (int off = 8; off; off >>= 1) mx[r] = fmaxf(mx[r], __shfl_xor(mx[r], off, 32));
    if (lm == 0) s_rv[lrow0 + r][wn] = mx[r];
  }
  __syncthreads();
  if (tid < FM)
    s_stat[tid] = fmaxf(fmaxf(s_rv[tid][0], s_rv[tid][1]),
                        fmaxf(s_rv[tid][2], s_rv[tid][3]));
  __syncthreads();

  // ---- pass 3: exp and row sum ----
  float sm[8];
#pragma unroll
  for (int r = 0; r < 8; r++) sm[r] = 0.f;
#pragma unroll
  for (int r = 0; r < 8; r++) {
    const float mr = s_stat[lrow0 + r];
#pragma unroll
    for (int j = 0; j < 16; j++) {
      const float e = __expf(acc[j][r] - mr);
      acc[j][r] = e;
      sm[r] += e;
    }
  }
#pragma unroll
  for (int r = 0; r < 8; r++) {
#pragma unroll
    for (int off = 8; off; off >>= 1) sm[r] += __shfl_xor(sm[r], off, 32);
    if (lm == 0) s_rv[lrow0 + r][wn] = sm[r];
  }
  __syncthreads();
  if (tid < FM)
    s_stat[tid] = 1.0f / (s_rv[tid][0] + s_rv[tid][1] + s_rv[tid][2] + s_rv[tid][3]);
  __syncthreads();

  // ---- pass 4: write normalized soft weights ----
#pragma unroll
  for (int r = 0; r < 8; r++) {
    const float inv = s_stat[lrow0 + r];
    float* wp = wsoft + (size_t)(m0 + lrow0 + r) * KCB + wn * 256 + lm;
#pragma unroll
    for (int j = 0; j < 16; j++) wp[j * 16] = acc[j][r] * inv;
  }

  // ---- pass 5: z_q = codebook[idx]; 4 rows per wave ----
#pragma unroll
  for (int q = 0; q < 4; q++) {
    const int lr = w * 4 + q;
    const int gi = s_idx[lr];
    const float* crow = codebook + (size_t)gi * DIM;
    float* zrow = zq + (size_t)(m0 + lr) * DIM;
#pragma unroll
    for (int t2 = 0; t2 < 16; t2++) zrow[t2 * 32 + lane] = crow[t2 * 32 + lane];
  }
}

// ---------------- launch ----------------
extern "C" void kernel_launch(void* const* d_in, const int* in_sizes, int n_in,
                              void* d_out, int out_size, void* d_ws, size_t ws_size,
                              hipStream_t stream) {
  (void)in_sizes; (void)n_in; (void)out_size; (void)ws_size;
  const float* state = (const float*)d_in[0];
  const float* adv   = (const float*)d_in[1];
  const float* cb    = (const float*)d_in[2];

  _Float16* mf16 = (_Float16*)d_ws;                 // [N_ROWS, DIM] f16 (32 MB)
  _Float16* cn16 = mf16 + (size_t)N_ROWS * DIM;     // [KCB,   DIM] f16 ( 1 MB)

  float* zq    = (float*)d_out;                     // [N_ROWS, DIM]
  float* wsoft = zq + (size_t)N_ROWS * DIM;         // [N_ROWS, KCB]
  float* idxo  = wsoft + (size_t)N_ROWS * KCB;      // [N_ROWS]

  mix_norm_kernel<<<N_ROWS / 8, 256, 0, stream>>>(state, adv, mf16);
  cb_norm_kernel<<<KCB / 8, 256, 0, stream>>>(cb, cn16);
  sim_gumbel_fused<<<N_ROWS / FM, 256, 0, stream>>>(mf16, cn16, cb, zq, wsoft, idxo);
}
